// Rnn_16690242912619
// MI455X (gfx1250) — matrix operand order repaired and checked by reading, not compiled, not run on hardware
//
#include <hip/hip_runtime.h>
#include <hip/hip_bf16.h>
#include <stdint.h>

#define L_SEQ 512
#define NB    256
#define IN0   256
#define HH    512
#define OUT_C 256

typedef __attribute__((ext_vector_type(16))) __bf16 bf16x16;
typedef __attribute__((ext_vector_type(8)))  __bf16 bf16x8;
typedef __attribute__((ext_vector_type(8)))  float  f32x8;

union BF16Frag { bf16x16 v; bf16x8 h[2]; };

__device__ inline unsigned short f2bf(float f) {
  union { float f; unsigned u; } v; v.f = f;
  unsigned r = v.u + 0x7FFFu + ((v.u >> 16) & 1u);   // round-to-nearest-even
  return (unsigned short)(r >> 16);
}

// A-fragment (16-bit A 16x32): per-lane chunks K=[kb,kb+8) and K=[kb+16,kb+24)
__device__ inline bf16x16 ld_a_frag(const __bf16* p) {
  BF16Frag u;
  u.h[0] = *(const bf16x8*)(p);
  u.h[1] = *(const bf16x8*)(p + 16);
  return u.v;
}
// B-fragment (16-bit B 32x16): per-lane chunks K=[kb,kb+8) and K=[kb+16,kb+24) of one column
__device__ inline bf16x16 ld_b_frag(const __bf16* p) {
  BF16Frag u;
  u.h[0] = *(const bf16x8*)(p);
  u.h[1] = *(const bf16x8*)(p + 16);
  return u.v;
}
__device__ inline f32x8 wmma_bf16(bf16x16 a, bf16x16 b, f32x8 c) {
  return __builtin_amdgcn_wmma_f32_16x16x32_bf16(false, a, false, b, (short)0, c,
                                                 false, false);
}
__device__ inline float sigmoid_f(float x) { return 1.0f / (1.0f + __expf(-x)); }
__device__ inline float tanh_f(float x)    { return 2.0f / (1.0f + __expf(-2.0f * x)) - 1.0f; }

// ---------------------------------------------------------------- prep kernels

__global__ void k_f32_to_bf16(const float* __restrict__ src,
                              unsigned short* __restrict__ dst, long n) {
  long i = ((long)blockIdx.x * blockDim.x + threadIdx.x) * 4;
  if (i < n) {
    float4 f = *(const float4*)(src + i);
    uint2 pk;
    pk.x = (unsigned)f2bf(f.x) | ((unsigned)f2bf(f.y) << 16);
    pk.y = (unsigned)f2bf(f.z) | ((unsigned)f2bf(f.w) << 16);
    *(uint2*)(dst + i) = pk;
  }
}

// Wcat[n][k] (row-major [1536][K0]): k<H -> Whh[n][k], else Wih[n][k-H]
__global__ void k_pack_wcat(const float* __restrict__ Whh, const float* __restrict__ Wih,
                            unsigned short* __restrict__ Wc, int K0, int Cin) {
  long i = (long)blockIdx.x * blockDim.x + threadIdx.x;
  long total = 1536L * K0;
  if (i < total) {
    int n = (int)(i / K0), k = (int)(i % K0);
    float v = (k < HH) ? Whh[(long)n * HH + k] : Wih[(long)n * Cin + (k - HH)];
    Wc[i] = f2bf(v);
  }
}

// bias layout: [ brz(1024)=bih+bhh | bnih(512)=bih[1024+c] | bnhh(512)=bhh[1024+c] ]
__global__ void k_pack_bias(const float* __restrict__ bih, const float* __restrict__ bhh,
                            float* __restrict__ out) {
  int i = blockIdx.x * blockDim.x + threadIdx.x;
  if (i < 1024)       out[i] = bih[i] + bhh[i];
  else if (i < 1536)  out[i] = bih[i];          // bnih
  else if (i < 2048)  out[i] = bhh[i - 512];    // bnhh
}

__global__ void k_init(const float* __restrict__ memory, float* __restrict__ h0buf,
                       float* __restrict__ h1buf, unsigned* __restrict__ ctr, int nctr) {
  int i = blockIdx.x * blockDim.x + threadIdx.x;
  if (i < NB * HH) {
    int n = i >> 9, c = i & 511;
    h0buf[i] = memory[n * 1024 + c];
    h1buf[i] = memory[n * 1024 + 512 + c];
  }
  if (i < nctr) ctr[i] = 0;
}

// ---------------------------------------------------------------- GRU scan
// Grid: 64 WGs x 256 threads. rowblock rb = wg>>2 (16 rows of batch), sub = wg&3.
// Each of 8 waves owns colgroup j = sub*8+wave (16 hidden cols) -> 4 gate tiles.
template <int K0, int CIN>
__global__ __launch_bounds__(256) void k_gru_scan(
    const unsigned short* __restrict__ xin,   // [L][NB][CIN] bf16
    const unsigned short* __restrict__ Wc,    // [1536][K0] bf16
    const float* __restrict__ bias,           // [2048]
    float* __restrict__ hbuf,                 // ping-pong 2*[NB][HH] f32
    unsigned short* __restrict__ yout,        // [L][NB][HH] bf16
    const unsigned char* __restrict__ done,   // [L][NB]
    unsigned* __restrict__ ctr)               // 16 counters, stride 32
{
  constexpr int AP = K0 + 8;                  // +16B row pad vs LDS banks
  __shared__ __attribute__((aligned(16))) unsigned short As[16 * AP];
  __shared__ unsigned char dbuf[16];

  const int tid  = threadIdx.x;
  const int lane = tid & 31;
  const int wave = tid >> 5;
  const int wg   = blockIdx.x;
  const int rb   = wg >> 2;
  const int sub  = wg & 3;
  const int row0 = rb * 16;
  const int j    = sub * 8 + wave;            // colgroup 0..31
  const int nloc = (j << 4) + (lane & 15);    // hidden column 0..511
  const int kb_a = (lane & 16) ? 8 : 0;       // A per-lane K sub-offset
  const int kb_b = (lane & 16) ? 8 : 0;       // B per-lane K sub-offset
  const int mbase = (lane & 16) ? 8 : 0;      // C/D row base per lane half

  const __bf16* WrowR = (const __bf16*)Wc + (size_t)nloc * K0;
  const __bf16* WrowZ = WrowR + (size_t)512 * K0;
  const __bf16* WrowN = WrowR + (size_t)1024 * K0;

  const float br  = bias[nloc];
  const float bz  = bias[512 + nloc];
  const float bni = bias[1024 + nloc];
  const float bnh = bias[1536 + nloc];

  unsigned* my_ctr = ctr + rb * 32;

  for (int t = 0; t < L_SEQ; ++t) {
    const float* hin  = hbuf + (size_t)(t & 1) * (NB * HH);
    float*       hout = hbuf + (size_t)((t + 1) & 1) * (NB * HH);

    // ---- stage A = [h(bf16) | x_t(bf16)] into LDS (rows row0..row0+15)
    for (int idx = tid; idx < 16 * (HH / 4); idx += 256) {
      int m = idx >> 7, k4 = idx & 127;
      float4 f = *(const float4*)(hin + (size_t)(row0 + m) * HH + (k4 << 2));
      uint2 pk;
      pk.x = (unsigned)f2bf(f.x) | ((unsigned)f2bf(f.y) << 16);
      pk.y = (unsigned)f2bf(f.z) | ((unsigned)f2bf(f.w) << 16);
      *(uint2*)&As[m * AP + (k4 << 2)] = pk;
    }
    for (int idx = tid; idx < 16 * (CIN / 4); idx += 256) {
      int m = idx / (CIN / 4), k4 = idx % (CIN / 4);
      uint2 v = *(const uint2*)(xin + ((size_t)t * NB + row0 + m) * CIN + (k4 << 2));
      *(uint2*)&As[m * AP + HH + (k4 << 2)] = v;
    }
    if (tid < 16) dbuf[tid] = done[(size_t)t * NB + row0 + tid];
    __syncthreads();

    // ---- WMMA: r,z fused over full K; n-gate split into h-part and x-part
    f32x8 c_r  = {0.f,0.f,0.f,0.f,0.f,0.f,0.f,0.f};
    f32x8 c_z  = {0.f,0.f,0.f,0.f,0.f,0.f,0.f,0.f};
    f32x8 c_hn = {0.f,0.f,0.f,0.f,0.f,0.f,0.f,0.f};
    f32x8 c_in = {0.f,0.f,0.f,0.f,0.f,0.f,0.f,0.f};
    const __bf16* Arow = (const __bf16*)&As[(lane & 15) * AP];

#pragma unroll 4
    for (int ks = 0; ks < HH / 32; ++ks) {            // h-part
      int k0 = ks * 32;
      bf16x16 a  = ld_a_frag(Arow + k0 + kb_a);
      bf16x16 brf = ld_b_frag(WrowR + k0 + kb_b);
      bf16x16 bzf = ld_b_frag(WrowZ + k0 + kb_b);
      bf16x16 bnf = ld_b_frag(WrowN + k0 + kb_b);
      c_r  = wmma_bf16(a, brf, c_r);
      c_z  = wmma_bf16(a, bzf, c_z);
      c_hn = wmma_bf16(a, bnf, c_hn);
    }
#pragma unroll 4
    for (int ks = HH / 32; ks < K0 / 32; ++ks) {      // x-part
      int k0 = ks * 32;
      bf16x16 a  = ld_a_frag(Arow + k0 + kb_a);
      bf16x16 brf = ld_b_frag(WrowR + k0 + kb_b);
      bf16x16 bzf = ld_b_frag(WrowZ + k0 + kb_b);
      bf16x16 bnf = ld_b_frag(WrowN + k0 + kb_b);
      c_r  = wmma_bf16(a, brf, c_r);
      c_z  = wmma_bf16(a, bzf, c_z);
      c_in = wmma_bf16(a, bnf, c_in);
    }

    // ---- gates + state update (C/D layout: elem i -> row mbase+i, col nloc)
#pragma unroll
    for (int i = 0; i < 8; ++i) {
      const int rl  = mbase + i;
      const int row = row0 + rl;
      const float r    = sigmoid_f(c_r[i] + br);
      const float z    = sigmoid_f(c_z[i] + bz);
      const float nn   = tanh_f(c_in[i] + bni + r * (c_hn[i] + bnh));
      const float hp   = hin[(size_t)row * HH + nloc];
      const float hnew = (1.0f - z) * nn + z * hp;
      yout[((size_t)t * NB + row) * HH + nloc] = f2bf(hnew);
      const float df = dbuf[rl] ? 1.0f : 0.0f;
      hout[(size_t)row * HH + nloc] = hnew * (1.0f - df);
    }

    // ---- per-rowblock barrier (4 WGs share a rowblock; h is ping-ponged)
    __syncthreads();
    __builtin_amdgcn_fence(__ATOMIC_RELEASE, "agent");
    if (tid == 0) {
      atomicAdd(my_ctr, 1u);
      const unsigned tgt = 4u * (unsigned)(t + 1);
      while (__hip_atomic_load(my_ctr, __ATOMIC_RELAXED, __HIP_MEMORY_SCOPE_AGENT) < tgt)
        __builtin_amdgcn_s_sleep(2);
    }
    __syncthreads();
    __builtin_amdgcn_fence(__ATOMIC_ACQUIRE, "agent");
  }
}

// ---------------------------------------------------------------- projection
// out[LN][256] = y1 @ Wp^T + bp.  8192 WGs x 128 thr; wave handles 4 col tiles.
__global__ __launch_bounds__(128) void k_proj(const unsigned short* __restrict__ y1,
                                              const unsigned short* __restrict__ WpB,
                                              const float* __restrict__ bp,
                                              float* __restrict__ out) {
  __shared__ __attribute__((aligned(16))) unsigned short As[16 * 520];
  const int tid = threadIdx.x, lane = tid & 31, wave = tid >> 5;
  const size_t row0 = (size_t)blockIdx.x * 16;
  for (int idx = tid; idx < 16 * 128; idx += 128) {
    int m = idx >> 7, k4 = idx & 127;
    uint2 v = *(const uint2*)(y1 + (row0 + m) * HH + (k4 << 2));
    *(uint2*)&As[m * 520 + (k4 << 2)] = v;
  }
  __syncthreads();
  const int kb_a = (lane & 16) ? 8 : 0;
  const int kb_b = (lane & 16) ? 8 : 0;
  const int mbase = (lane & 16) ? 8 : 0;
  const __bf16* Arow = (const __bf16*)&As[(lane & 15) * 520];
#pragma unroll
  for (int jj = 0; jj < 4; ++jj) {
    const int j = wave * 4 + jj;
    const int n = (j << 4) + (lane & 15);
    const __bf16* Wr = (const __bf16*)WpB + (size_t)n * HH;
    f32x8 c = {0.f,0.f,0.f,0.f,0.f,0.f,0.f,0.f};
#pragma unroll 4
    for (int ks = 0; ks < 16; ++ks) {
      bf16x16 a = ld_a_frag(Arow + ks * 32 + kb_a);
      bf16x16 b = ld_b_frag(Wr + ks * 32 + kb_b);
      c = wmma_bf16(a, b, c);
    }
    const float bias = bp[n];
#pragma unroll
    for (int i = 0; i < 8; ++i)
      out[(row0 + mbase + i) * OUT_C + n] = c[i] + bias;
  }
}

__global__ void k_hn(const float* __restrict__ h0, const float* __restrict__ h1,
                     float* __restrict__ outp) {
  int i = blockIdx.x * blockDim.x + threadIdx.x;
  if (i < NB * HH) {
    int n = i >> 9, c = i & 511;
    outp[n * 1024 + c]       = h0[i];
    outp[n * 1024 + 512 + c] = h1[i];
  }
}

// ---------------------------------------------------------------- launch

extern "C" void kernel_launch(void* const* d_in, const int* in_sizes, int n_in,
                              void* d_out, int out_size, void* d_ws, size_t ws_size,
                              hipStream_t stream) {
  const float* x      = (const float*)d_in[0];
  const float* memory = (const float*)d_in[1];
  const unsigned char* done = (const unsigned char*)d_in[2];
  const float* Wih0 = (const float*)d_in[3];
  const float* Whh0 = (const float*)d_in[4];
  const float* bih0 = (const float*)d_in[5];
  const float* bhh0 = (const float*)d_in[6];
  const float* Wih1 = (const float*)d_in[7];
  const float* Whh1 = (const float*)d_in[8];
  const float* bih1 = (const float*)d_in[9];
  const float* bhh1 = (const float*)d_in[10];
  const float* Wp   = (const float*)d_in[11];
  const float* bp   = (const float*)d_in[12];

  char* w = (char*)d_ws;
  size_t o = 0;
  auto alloc = [&](size_t bytes) -> void* {
    void* p = w + o;
    o = (o + bytes + 255) & ~(size_t)255;
    return p;
  };
  unsigned short* x_bf  = (unsigned short*)alloc((size_t)L_SEQ * NB * IN0 * 2); // 64 MB
  unsigned short* y0_bf = (unsigned short*)alloc((size_t)L_SEQ * NB * HH * 2);  // 128 MB
  unsigned short* y1_bf = (unsigned short*)alloc((size_t)L_SEQ * NB * HH * 2);  // 128 MB
  unsigned short* Wc0   = (unsigned short*)alloc((size_t)1536 * 768 * 2);
  unsigned short* Wc1   = (unsigned short*)alloc((size_t)1536 * 1024 * 2);
  unsigned short* Wp_bf = (unsigned short*)alloc((size_t)OUT_C * HH * 2);
  float* bias0 = (float*)alloc(2048 * 4);
  float* bias1 = (float*)alloc(2048 * 4);
  float* h0buf = (float*)alloc((size_t)2 * NB * HH * 4);
  float* h1buf = (float*)alloc((size_t)2 * NB * HH * 4);
  unsigned* ctr = (unsigned*)alloc(1024 * 4);   // 2 layers x 16 rb x stride 32

  // prep
  {
    long n = (long)L_SEQ * NB * IN0;
    k_f32_to_bf16<<<(unsigned)((n / 4 + 255) / 256), 256, 0, stream>>>(x, x_bf, n);
  }
  {
    long n = (long)OUT_C * HH;
    k_f32_to_bf16<<<(unsigned)((n / 4 + 255) / 256), 256, 0, stream>>>(Wp, Wp_bf, n);
  }
  k_pack_wcat<<<(1536 * 768 + 255) / 256, 256, 0, stream>>>(Whh0, Wih0, Wc0, 768, IN0);
  k_pack_wcat<<<(1536 * 1024 + 255) / 256, 256, 0, stream>>>(Whh1, Wih1, Wc1, 1024, HH);
  k_pack_bias<<<8, 256, 0, stream>>>(bih0, bhh0, bias0);
  k_pack_bias<<<8, 256, 0, stream>>>(bih1, bhh1, bias1);
  k_init<<<(NB * HH + 255) / 256, 256, 0, stream>>>(memory, h0buf, h1buf, ctr, 1024);

  // sequential GRU scans (persistent WMMA kernels)
  k_gru_scan<768, IN0><<<64, 256, 0, stream>>>(x_bf, Wc0, bias0, h0buf, y0_bf, done, ctr);
  k_gru_scan<1024, HH><<<64, 256, 0, stream>>>(y0_bf, Wc1, bias1, h1buf, y1_bf, done,
                                               ctr + 512);

  // projection + hn
  float* out = (float*)d_out;
  k_proj<<<(L_SEQ * NB) / 16, 128, 0, stream>>>(y1_bf, Wp_bf, bp, out);
  k_hn<<<(NB * HH + 255) / 256, 256, 0, stream>>>(h0buf, h1buf,
                                                  out + (size_t)L_SEQ * NB * OUT_C);
}
